// Attention_83562883711706
// MI455X (gfx1250) — matrix operand order repaired and checked by reading, not compiled, not run on hardware
//
#include <hip/hip_runtime.h>
#include <cstdint>
#include <cstddef>

// ---------------------------------------------------------------------------
// MI455X (gfx1250) seq2seq RNN + attention.
// ~98% of FLOPs are 4 GEMMs of form A[M,K] @ B[N,K]^T -> v_wmma_f32_16x16x32_bf16
// (f32 accum). Both operands are K-contiguous, matching CDNA5 WMMA A/B VGPR
// layouts directly. B tiles are DMA'd into a double-buffered LDS tile with
// GLOBAL_LOAD_ASYNC_TO_LDS_B128 (ASYNCcnt + s_wait_asynccnt), converted to
// bf16 at fragment-read time with native __bf16 casts.
// Sequential RNN scans run as single-workgroup kernels with h in LDS.
// ---------------------------------------------------------------------------

typedef __attribute__((ext_vector_type(16))) __bf16 v16bf;
typedef __attribute__((ext_vector_type(8)))  float  v8f;

__device__ __forceinline__ v16bf cvt16(float4 a0, float4 a1, float4 a2, float4 a3) {
  v16bf r;
  r[0]  = (__bf16)a0.x; r[1]  = (__bf16)a0.y; r[2]  = (__bf16)a0.z; r[3]  = (__bf16)a0.w;
  r[4]  = (__bf16)a1.x; r[5]  = (__bf16)a1.y; r[6]  = (__bf16)a1.z; r[7]  = (__bf16)a1.w;
  r[8]  = (__bf16)a2.x; r[9]  = (__bf16)a2.y; r[10] = (__bf16)a2.z; r[11] = (__bf16)a2.w;
  r[12] = (__bf16)a3.x; r[13] = (__bf16)a3.y; r[14] = (__bf16)a3.z; r[15] = (__bf16)a3.w;
  return r;
}

// 64B async DMA: global (saddr + 32-bit voffset) -> LDS (32-bit lds addr).
// INST_OFFSET applies to both global and LDS addresses (ISA 08 §4.4).
__device__ __forceinline__ void async_copy64B(uint32_t ldsAddr, uint32_t gOff,
                                              const float* base) {
  asm volatile(
      "global_load_async_to_lds_b128 %0, %1, %2 offset:0\n\t"
      "global_load_async_to_lds_b128 %0, %1, %2 offset:16\n\t"
      "global_load_async_to_lds_b128 %0, %1, %2 offset:32\n\t"
      "global_load_async_to_lds_b128 %0, %1, %2 offset:48"
      :
      : "v"(ldsAddr), "v"(gOff), "s"(base)
      : "memory");
}

__device__ __forceinline__ void wait_async0() {
  asm volatile("s_wait_asynccnt 0x0" ::: "memory");
}

#define TM 128
#define TN 128
#define TK 32

// D[M,N] = A[M,K] * B[N,K]^T + bias1[n] + bias2[n]
__global__ __launch_bounds__(256) void gemm_bf16_wmma(
    const float* __restrict__ A, const float* __restrict__ B,
    float* __restrict__ D, int M, int N, int K,
    const float* __restrict__ bias1, const float* __restrict__ bias2)
{
  __shared__ float ldsBf[2][TN * TK];        // 2 x 16 KB double buffer, f32

  const int tid  = threadIdx.x;
  const int lane = tid & 31;
  const int wave = tid >> 5;                 // 0..7
  const int lrow = lane & 15;
  const int hh   = lane >> 4;                // lane half (wave32)

  const int mBlk = blockIdx.y * TM;
  const int nBlk = blockIdx.x * TN;
  const int mWave = (wave >> 1) * 32;        // 4 waves along M
  const int nWave = (wave & 1) * 64;         // 2 waves along N

  const v8f vzero = {0.f,0.f,0.f,0.f,0.f,0.f,0.f,0.f};
  v8f acc[2][4];
  #pragma unroll
  for (int i = 0; i < 2; ++i)
    #pragma unroll
    for (int j = 0; j < 4; ++j) acc[i][j] = vzero;

  // async staging map: thread -> (n = tid/2, 16-float k-chunk = 64 bytes)
  const int sn = tid >> 1;                   // 0..127
  const int sk = (tid & 1) * 16;             // 0 or 16
  uint32_t gOff = (uint32_t)(((uint32_t)(nBlk + sn) * (uint32_t)K + (uint32_t)sk) * 4u);
  const uint32_t gStep = (uint32_t)(TK * 4);
  uint32_t lA[2];
  lA[0] = (uint32_t)(uintptr_t)&ldsBf[0][sn * TK + sk];   // low 32 bits give the LDS byte address
  lA[1] = (uint32_t)(uintptr_t)&ldsBf[1][sn * TK + sk];

  const float* Arow[2];
  Arow[0] = A + (size_t)(mBlk + mWave +  0 + lrow) * K;
  Arow[1] = A + (size_t)(mBlk + mWave + 16 + lrow) * K;

  const int nsteps = K / TK;

  // prologue: DMA tile 0 into buffer 0
  async_copy64B(lA[0], gOff, B);

  for (int step = 0; step < nsteps; ++step) {
    const int k0  = step * TK;
    const int buf = step & 1;

    wait_async0();          // this wave's DMA for tile `step` landed in LDS
    __syncthreads();        // all waves' DMA done; all reads of buf^1 finished

    if (step + 1 < nsteps) {
      async_copy64B(lA[buf ^ 1], gOff + gStep, B);   // overlap next tile DMA
      __builtin_prefetch(Arow[0] + k0 + TK);          // global_prefetch_b8
    }
    gOff += gStep;

    // ---- A fragments straight from global (lane m = lrow; K split 8/8
    //      across lane halves per ISA 16-bit A layout) ----
    v16bf afr[2];
    #pragma unroll
    for (int mf = 0; mf < 2; ++mf) {
      const int c0 = k0 + 8 * hh;        // K = 8h + e      (e = 0..7)
      const int c1 = k0 + 16 + 8 * hh;   // K = 16 + 8h + e (e = 8..15)
      float4 a0 = *(const float4*)(Arow[mf] + c0);
      float4 a1 = *(const float4*)(Arow[mf] + c0 + 4);
      float4 a2 = *(const float4*)(Arow[mf] + c1);
      float4 a3 = *(const float4*)(Arow[mf] + c1 + 4);
      afr[mf] = cvt16(a0, a1, a2, a3);
    }

    // ---- B fragments from LDS f32 (lane n = lrow; elements K = 8h+e, 16+8h+e) ----
    v16bf bfr[4];
    #pragma unroll
    for (int nf = 0; nf < 4; ++nf) {
      const int nl = nWave + nf * 16 + lrow;
      const float* bp = &ldsBf[buf][nl * TK + 8 * hh];
      float4 b0 = *(const float4*)(bp + 0);
      float4 b1 = *(const float4*)(bp + 4);
      float4 b2 = *(const float4*)(bp + 16);
      float4 b3 = *(const float4*)(bp + 20);
      bfr[nf] = cvt16(b0, b1, b2, b3);
    }

    // ---- 8x v_wmma_f32_16x16x32_bf16 per wave per K-step ----
    #pragma unroll
    for (int mf = 0; mf < 2; ++mf)
      #pragma unroll
      for (int nf = 0; nf < 4; ++nf)
        acc[mf][nf] = __builtin_amdgcn_wmma_f32_16x16x32_bf16(
            false, afr[mf], false, bfr[nf], (short)0, acc[mf][nf], false, false);
  }

  // ---- epilogue: C/D layout lane n = lrow, VGPR v -> m = v + 8*half ----
  #pragma unroll
  for (int mf = 0; mf < 2; ++mf)
    #pragma unroll
    for (int nf = 0; nf < 4; ++nf) {
      const int ng = nBlk + nWave + nf * 16 + lrow;
      float bv = (bias1 ? bias1[ng] : 0.f) + (bias2 ? bias2[ng] : 0.f);
      #pragma unroll
      for (int v = 0; v < 8; ++v) {
        const int mg = mBlk + mWave + mf * 16 + 8 * hh + v;
        D[(size_t)mg * N + ng] = acc[mf][nf][v] + bv;
      }
    }
}

// out[c][r] = in[r][c]
__global__ __launch_bounds__(256) void transpose_f32(
    const float* __restrict__ in, float* __restrict__ out, int rows, int cols)
{
  __shared__ float tile[32][33];
  const int bx = blockIdx.x * 32;       // col base
  const int by = blockIdx.y * 32;       // row base
  const int tx = threadIdx.x & 31;
  const int ty = threadIdx.x >> 5;      // 0..7
  #pragma unroll
  for (int i = 0; i < 32; i += 8)
    tile[ty + i][tx] = in[(size_t)(by + ty + i) * cols + bx + tx];
  __syncthreads();
  #pragma unroll
  for (int i = 0; i < 32; i += 8)
    out[(size_t)(bx + ty + i) * rows + by + tx] = tile[tx][ty + i];
}

// Sequential encoder scan: h_t = tanh(Z[t] + Whh h), biases prefolded in Z.
__global__ __launch_bounds__(512) void enc_rnn_scan(
    const float* __restrict__ Z,       // [S,512]
    const float* __restrict__ WhhT,    // [512,512] transposed recurrent weight
    const float* __restrict__ h0,      // [512]
    float* __restrict__ hs,            // [S,512]
    int S)
{
  __shared__ float h[512];
  const int i = threadIdx.x;
  h[i] = h0[i];
  __syncthreads();
  for (int t = 0; t < S; ++t) {
    float acc = Z[(size_t)t * 512 + i];
    #pragma unroll 8
    for (int j = 0; j < 512; ++j)                 // coalesced rows, LDS broadcast
      acc += WhhT[(size_t)j * 512 + i] * h[j];
    const float hn = tanhf(acc);
    __syncthreads();
    h[i] = hn;
    hs[(size_t)t * 512 + i] = hn;
    __syncthreads();
  }
}

// Decoder: recurrence + attention scores + softmax + context, sequential in t.
__global__ __launch_bounds__(1024) void dec_attn_scan(
    const float* __restrict__ Zdec,      // [256,512] prebias
    const float* __restrict__ WhhT,      // [512,512]
    const float* __restrict__ attn_encT, // [512,1024]
    const float* __restrict__ enc_hs,    // [1024,512]
    const float* __restrict__ hinit,     // [512] = enc_hs last row
    float* __restrict__ cat,             // [256,1024] = [h | context]
    float* __restrict__ attn_out,        // [256,1024]
    int Sdec, int Senc)
{
  __shared__ float h[512];
  __shared__ float w[1024];
  __shared__ float red[1024];
  __shared__ float ctx[1024];
  const int tid = threadIdx.x;
  if (tid < 512) h[tid] = hinit[tid];
  __syncthreads();

  for (int t = 0; t < Sdec; ++t) {
    // h = tanh(Z + Whh h)
    float hn = 0.f;
    if (tid < 512) {
      float acc = Zdec[(size_t)t * 512 + tid];
      #pragma unroll 8
      for (int j = 0; j < 512; ++j)
        acc += WhhT[(size_t)j * 512 + tid] * h[j];
      hn = tanhf(acc);
    }
    __syncthreads();
    if (tid < 512) { h[tid] = hn; cat[(size_t)t * 1024 + tid] = hn; }
    __syncthreads();

    // scores: s = tid over S_enc = blockDim = 1024
    float sc = 0.f;
    #pragma unroll 8
    for (int j = 0; j < 512; ++j)
      sc += attn_encT[(size_t)j * Senc + tid] * h[j];

    // softmax over 1024 (LDS tree reductions)
    red[tid] = sc; __syncthreads();
    for (int off = 512; off > 0; off >>= 1) {
      if (tid < off) red[tid] = fmaxf(red[tid], red[tid + off]);
      __syncthreads();
    }
    const float mx = red[0]; __syncthreads();
    const float e = __expf(sc - mx);
    red[tid] = e; __syncthreads();
    for (int off = 512; off > 0; off >>= 1) {
      if (tid < off) red[tid] += red[tid + off];
      __syncthreads();
    }
    const float inv = 1.f / red[0]; __syncthreads();
    const float wn = e * inv;
    w[tid] = wn;
    attn_out[(size_t)t * Senc + tid] = wn;
    __syncthreads();

    // context[j] = sum_s w[s] * enc_hs[s][j]; split s-range over 2 halves
    const int j  = tid & 511;
    const int sg = tid >> 9;
    float c = 0.f;
    for (int s = sg * 512; s < sg * 512 + 512; ++s)   // coalesced across j
      c += w[s] * enc_hs[(size_t)s * 512 + j];
    ctx[tid] = c; __syncthreads();
    if (tid < 512) cat[(size_t)t * 1024 + 512 + tid] = ctx[tid] + ctx[512 + tid];
    __syncthreads();
  }
}

extern "C" void kernel_launch(void* const* d_in, const int* in_sizes, int n_in,
                              void* d_out, int out_size, void* d_ws, size_t ws_size,
                              hipStream_t stream) {
  constexpr int C = 32000, H = 512, SE = 1024, SD = 256;

  const float* x_enc   = (const float*)d_in[0];   // [1024,32000]
  const float* h0      = (const float*)d_in[1];   // [512]
  const float* x_dec   = (const float*)d_in[2];   // [256,32000]
  const float* enc_Wih = (const float*)d_in[3];   // [512,32000]
  const float* enc_Whh = (const float*)d_in[4];   // [512,512]
  const float* enc_bih = (const float*)d_in[5];
  const float* enc_bhh = (const float*)d_in[6];
  const float* dec_Wih = (const float*)d_in[7];
  const float* dec_Whh = (const float*)d_in[8];
  const float* dec_bih = (const float*)d_in[9];
  const float* dec_bhh = (const float*)d_in[10];
  const float* attn_W  = (const float*)d_in[11];  // [512,512]
  const float* attn_b  = (const float*)d_in[12];
  const float* out_W   = (const float*)d_in[13];  // [32000,1024]
  const float* out_b   = (const float*)d_in[14];

  float* out      = (float*)d_out;                // [256,32000]
  float* attn_out = out + (size_t)SD * C;         // [256,1024]

  float* ws = (float*)d_ws;
  float* Zenc     = ws;               ws += (size_t)SE * H;   // 1024x512
  float* Zdec     = ws;               ws += (size_t)SD * H;   // 256x512
  float* enc_hs   = ws;               ws += (size_t)SE * H;   // 1024x512
  float* attn_enc = ws;               ws += (size_t)SE * H;   // 1024x512
  float* attn_encT= ws;               ws += (size_t)SE * H;   // 512x1024
  float* WhhT_e   = ws;               ws += (size_t)H * H;
  float* WhhT_d   = ws;               ws += (size_t)H * H;
  float* cat      = ws;               ws += (size_t)SD * 2 * H; // 256x1024
  (void)ws_size; (void)n_in; (void)in_sizes; (void)out_size;

  // 1) transpose recurrent weights for coalesced sequential matvecs
  transpose_f32<<<dim3(H / 32, H / 32), 256, 0, stream>>>(enc_Whh, WhhT_e, H, H);
  transpose_f32<<<dim3(H / 32, H / 32), 256, 0, stream>>>(dec_Whh, WhhT_d, H, H);

  // 2) Z_enc = X_enc @ enc_Wih^T + bih + bhh   (M=1024,N=512,K=32000)
  gemm_bf16_wmma<<<dim3(H / TN, SE / TM), 256, 0, stream>>>(
      x_enc, enc_Wih, Zenc, SE, H, C, enc_bih, enc_bhh);

  // 3) Z_dec = X_dec @ dec_Wih^T + bih + bhh   (M=256,N=512,K=32000)
  gemm_bf16_wmma<<<dim3(H / TN, SD / TM), 256, 0, stream>>>(
      x_dec, dec_Wih, Zdec, SD, H, C, dec_bih, dec_bhh);

  // 4) encoder scan (sequential)
  enc_rnn_scan<<<1, 512, 0, stream>>>(Zenc, WhhT_e, h0, enc_hs, SE);

  // 5) attn_enc = enc_hs @ attn_W^T + attn_b   (M=1024,N=512,K=512)
  gemm_bf16_wmma<<<dim3(H / TN, SE / TM), 256, 0, stream>>>(
      enc_hs, attn_W, attn_enc, SE, H, H, attn_b, nullptr);
  transpose_f32<<<dim3(H / 32, SE / 32), 256, 0, stream>>>(attn_enc, attn_encT, SE, H);

  // 6) decoder scan: recurrence + softmax attention + context
  dec_attn_scan<<<1, 1024, 0, stream>>>(
      Zdec, WhhT_d, attn_encT, enc_hs, enc_hs + (size_t)(SE - 1) * H,
      cat, attn_out, SD, SE);

  // 7) outputs = cat @ out_W^T + out_b   (M=256,N=32000,K=1024)
  gemm_bf16_wmma<<<dim3(C / TN, SD / TM), 256, 0, stream>>>(
      cat, out_W, out, SD, C, 2 * H, out_b, nullptr);
}
